// PairwiseDualPatchCNNCAViT_62027917689508
// MI455X (gfx1250) — matrix operand order repaired and checked by reading, not compiled, not run on hardware
//
#include <hip/hip_runtime.h>
#include <hip/hip_bf16.h>

// ---------------------------------------------------------------------------
// Shapes
// ---------------------------------------------------------------------------
#define Cc     256
#define HWs    64
#define SP     4096                 // 64*64 spatial per image
#define BATCH  16
#define IMG    (Cc * SP)            // elems per image  = 1,048,576
#define TENS   (BATCH * IMG)        // elems per tensor = 16,777,216

typedef __bf16 bf16_t;
typedef bf16_t v16bf __attribute__((ext_vector_type(16)));
typedef bf16_t v8bf  __attribute__((ext_vector_type(8)));
typedef float  v8f   __attribute__((ext_vector_type(8)));

#define ASPLANE (68 * 32)           // padded A slab per row: x in -2..65, 32 ch
#define TSTRIDE 68                  // padded f32 transpose tile stride

__device__ __forceinline__ float lrelu_f(float v) { return v > 0.f ? v : 0.2f * v; }

// ---------------------------------------------------------------------------
// Weight repack: fp32 OIHW -> bf16 in *WMMA B-fragment order* so the GEMM can
// load each lane's 16 bf16 B values as one contiguous 32-byte global load.
//
// Fragment index:  idx = (((kk*4 + g)*4 + ns)*32 + lane)*16 + t
//   kk   : K chunk (kpos*8 + cchunk), K = kk*32 + (lane>>4)*16 + t
//   g    : output-channel group (64 wide), ns : 16-wide N subtile
//   lane : B layout -> col = g*64 + ns*16 + (lane&15), K-half = lane>>4
// ---------------------------------------------------------------------------
__global__ void repack_w(const float* __restrict__ w, bf16_t* __restrict__ out, int nkpos) {
    int idx = blockIdx.x * 256 + threadIdx.x;
    int total = nkpos * Cc * Cc;
    if (idx >= total) return;
    int t    = idx & 15;
    int l    = (idx >> 4) & 31;
    int ns   = (idx >> 9) & 3;
    int g    = (idx >> 11) & 3;
    int kk   = idx >> 13;
    int o    = g * 64 + ns * 16 + (l & 15);
    int kgl  = kk * 32 + (l >> 4) * 16 + t;   // kpos-major global K
    int kpos = kgl >> 8;
    int c    = kgl & 255;
    float v = (nkpos == 1) ? w[o * Cc + c] : w[(o * Cc + c) * 9 + kpos];
    out[idx] = (bf16_t)v;
}

// ---------------------------------------------------------------------------
// Implicit-GEMM conv, templated on tap count / dilation so tap loops unroll.
// Block = 128 threads (4 waves).  Block tile: 128 spatial (two image rows
// y0, y0+1) x 64 out channels.  Wave w owns x in [16w,16w+16) of BOTH rows:
// 8 accumulators (2 m-subtiles x 4 n-subtiles); each B fragment feeds 2 WMMAs.
//
// Per (ky, c-chunk) stage: padded bf16 slabs As[2][68][32] staged once and
// reused by all three kx taps.  Staging: per thread one x-pair x 8 channels ->
// clause of global_load_b64 at constant offsets + two ds_store_b128.
// A fragments: two ds_load_b128 per lane (contiguous K order as in B: K runs
// [16h,16h+8) and [16h+8,16h+16), k = 16h + s).  B fragments: 32-byte
// coalesced global loads from fragment-ordered weights.
// Epilogue: LDS transpose to padded T[32][68] then coalesced float4
// bias -> preBN -> lrelu -> residual(+gamma) -> postBN -> store.
// ---------------------------------------------------------------------------
template <int NKPOS, int DIL>
__global__ __launch_bounds__(128) void conv_gemm(
    const float* __restrict__ in,      // [16,256,64,64]
    const bf16_t* __restrict__ Bw,     // fragment-ordered bf16 weights
    const float* __restrict__ bias,
    const float* __restrict__ bnw, const float* __restrict__ bnb,   // pre-BN
    int do_lrelu,
    const float* __restrict__ resid,
    const float* __restrict__ gammap,  // device scalar gate (null -> 1.0)
    const float* __restrict__ pbw, const float* __restrict__ pbb,   // post-BN
    float* __restrict__ out)
{
    constexpr int NKY = (NKPOS == 9) ? 3 : 1;
    constexpr int NKX = (NKPOS == 9) ? 3 : 1;

    const int b  = blockIdx.x >> 5;
    const int y0 = (blockIdx.x & 31) << 1;   // two rows: y0, y0+1
    const int g  = blockIdx.y;               // output-channel group (64 wide)
    const int og = g * 64;
    const int tid  = threadIdx.x;
    const int wave = tid >> 5;
    const int lane = tid & 31;
    const int lrow = lane & 15;
    const int half = lane >> 4;

    __shared__ alignas(16) unsigned char smem[2 * ASPLANE * 2];  // 8704 B
    bf16_t* As = (bf16_t*)smem;              // [2][68][32] during K loop
    float*  T  = (float*)smem;               // [32][68] f32 during epilogue

    v8f acc[2][4];
    #pragma unroll
    for (int m = 0; m < 2; ++m)
        #pragma unroll
        for (int ns = 0; ns < 4; ++ns)
            #pragma unroll
            for (int r = 0; r < 8; ++r) acc[m][ns][r] = 0.f;

    const float* inb = in + (size_t)b * IMG;
    const v16bf* Bf  = (const v16bf*)Bw;

    // Zero the horizontal halo once (padded x = {0,1,66,67}); main staging
    // never overwrites it, and T-phase only starts after the K loop.
    {
        int c2 = (tid & 15) * 2;
        int hx = (tid >> 4) & 3;
        int ry = tid >> 6;
        int xx = (hx < 2) ? hx : (64 + hx);
        *(uint32_t*)&As[ry * ASPLANE + xx * 32 + c2] = 0u;
    }

    const int xp    = (tid & 31) * 2;        // x pair this thread stages
    const int cbase = (tid >> 5) * 8;        // 8 channels this thread stages

    for (int s = 0; s < NKY * 8; ++s) {
        const int ky = s >> 3;
        const int cc = s & 7;
        const int c0 = cc << 5;
        const int dy = (NKPOS == 9) ? (ky - 1) * DIL : 0;

        // Stage two row-slabs (uniform branch per row)
        #pragma unroll
        for (int ry = 0; ry < 2; ++ry) {
            const int yy = y0 + ry + dy;
            v8bf h0, h1;
            if (yy >= 0 && yy < 64) {
                const float* gp = inb + (size_t)(c0 + cbase) * SP + yy * 64 + xp;
                #pragma unroll
                for (int i = 0; i < 8; ++i) {
                    float2 f = *(const float2*)(gp + (size_t)i * SP);
                    h0[i] = (bf16_t)f.x;
                    h1[i] = (bf16_t)f.y;
                }
            } else {
                #pragma unroll
                for (int i = 0; i < 8; ++i) { h0[i] = (bf16_t)0.f; h1[i] = (bf16_t)0.f; }
            }
            *(v8bf*)&As[ry * ASPLANE + (xp + 2) * 32 + cbase] = h0;
            *(v8bf*)&As[ry * ASPLANE + (xp + 3) * 32 + cbase] = h1;
        }
        __syncthreads();

        #pragma unroll
        for (int kx = 0; kx < NKX; ++kx) {
            const int dx    = (NKPOS == 9) ? (kx - 1) * DIL : 0;
            const int kpos  = ky * 3 + kx;
            const int kkidx = kpos * 8 + cc;

            v16bf af[2];
            #pragma unroll
            for (int m = 0; m < 2; ++m) {
                const bf16_t* ar = &As[m * ASPLANE + (wave * 16 + lrow + dx + 2) * 32];
                v8bf lo = *(const v8bf*)(ar + half * 16);
                v8bf hi = *(const v8bf*)(ar + half * 16 + 8);
                af[m] = __builtin_shufflevector(lo, hi,
                    0, 1, 2, 3, 4, 5, 6, 7, 8, 9, 10, 11, 12, 13, 14, 15);
            }

            const size_t fb = (((size_t)kkidx * 4 + g) * 4) * 32 + lane;
            #pragma unroll
            for (int ns = 0; ns < 4; ++ns) {
                v16bf bfrag = Bf[fb + ns * 32];
                acc[0][ns] = __builtin_amdgcn_wmma_f32_16x16x32_bf16(
                    false, af[0], false, bfrag, (short)0, acc[0][ns], false, false);
                acc[1][ns] = __builtin_amdgcn_wmma_f32_16x16x32_bf16(
                    false, af[1], false, bfrag, (short)0, acc[1][ns], false, false);
            }
        }
        __syncthreads();
    }

    // ---------------- Epilogue: LDS transpose + coalesced stores ----------
    const float inv = 0.99999500003749969f;  // 1/sqrt(1+1e-5)
    const float gv  = resid ? (gammap ? gammap[0] : 1.0f) : 0.f;

    #pragma unroll
    for (int m = 0; m < 2; ++m) {
        const int yrow = y0 + m;
        #pragma unroll
        for (int oh = 0; oh < 2; ++oh) {
            // Waves deposit 2 n-subtiles (32 channels) into padded T[32][68].
            #pragma unroll
            for (int nsl = 0; nsl < 2; ++nsl) {
                const int ns = oh * 2 + nsl;
                const int ol = nsl * 16 + lrow;
                #pragma unroll
                for (int r = 0; r < 8; ++r) {
                    int x = wave * 16 + r + half * 8;
                    T[ol * TSTRIDE + x] = acc[m][ns][r];
                }
            }
            __syncthreads();
            // Cooperative epilogue: thread owns (channel, 16-x segment).
            const int ol  = tid >> 2;
            const int x0  = (tid & 3) * 16;
            const int o   = og + oh * 32 + ol;
            const float bi  = bias[o];
            const float w1f = bnw ? bnw[o] * inv : 1.f;
            const float b1f = bnb ? bnb[o] : 0.f;
            const float w2f = pbw ? pbw[o] * inv : 1.f;
            const float b2f = pbb ? pbb[o] : 0.f;
            const size_t obase = ((size_t)b * Cc + o) * SP + yrow * 64 + x0;
            #pragma unroll
            for (int i = 0; i < 16; i += 4) {
                float4 vv = *(const float4*)&T[ol * TSTRIDE + x0 + i];
                float4 rs = resid ? *(const float4*)&resid[obase + i]
                                  : make_float4(0.f, 0.f, 0.f, 0.f);
                float e[4] = {vv.x, vv.y, vv.z, vv.w};
                float rr[4] = {rs.x, rs.y, rs.z, rs.w};
                #pragma unroll
                for (int j = 0; j < 4; ++j) {
                    float v = e[j] + bi;
                    if (bnw) v = v * w1f + b1f;
                    if (do_lrelu) v = lrelu_f(v);
                    if (resid) v = rr[j] + gv * v;
                    if (pbw) v = v * w2f + b2f;
                    e[j] = v;
                }
                *(float4*)&out[obase + i] = make_float4(e[0], e[1], e[2], e[3]);
            }
            __syncthreads();
        }
    }
}

// ---------------------------------------------------------------------------
// Attention group 0 (patch 64x64 == whole image -> softmax over one logit is
// identity) -> Y channels [0,64) = V channels [0,64).
// ---------------------------------------------------------------------------
__global__ void att_copy(const float* __restrict__ v, float* __restrict__ y) {
    size_t i = (size_t)blockIdx.x * 256 + threadIdx.x;   // over 16*64*4096
    size_t b = i / (64 * SP);
    size_t r = i - b * (64 * SP);
    size_t src = b * IMG + r;
    y[src] = v[src];
}

// ---------------------------------------------------------------------------
// Attention scores: one block per (b, n, m): E-long dot * 1/sqrt(E).
// ---------------------------------------------------------------------------
__global__ void att_scores(const float* __restrict__ q, const float* __restrict__ k,
                           float* __restrict__ scores, int grp, int p) {
    const int ow = 64 / p, nw = ow * ow, pp = p * p, E = 64 * pp;
    int idx = blockIdx.x;
    const int m = idx % nw;  idx /= nw;
    const int n = idx % nw;  idx /= nw;
    const int b = idx;
    const int ny = (n / ow) * p, nx = (n % ow) * p;
    const int my = (m / ow) * p, mx = (m % ow) * p;
    const float* qb = q + ((size_t)b * Cc + grp * 64) * SP;
    const float* kb = k + ((size_t)b * Cc + grp * 64) * SP;
    float s = 0.f;
    for (int e = threadIdx.x; e < E; e += 256) {
        int d = e / pp; int r = e - d * pp; int py = r / p; int px = r - py * p;
        float qv = qb[(size_t)d * SP + (ny + py) * 64 + (nx + px)];
        float kv = kb[(size_t)d * SP + (my + py) * 64 + (mx + px)];
        s += qv * kv;
    }
    __shared__ float red[256];
    red[threadIdx.x] = s;
    __syncthreads();
    for (int off = 128; off > 0; off >>= 1) {
        if (threadIdx.x < off) red[threadIdx.x] += red[threadIdx.x + off];
        __syncthreads();
    }
    if (threadIdx.x == 0)
        scores[blockIdx.x] = red[0] * rsqrtf((float)E);
}

// ---------------------------------------------------------------------------
// Attention softmax + weighted sum of V windows -> Y.
// ---------------------------------------------------------------------------
__global__ void att_out(const float* __restrict__ v, const float* __restrict__ scores,
                        float* __restrict__ y, int grp, int p) {
    const int ow = 64 / p, nw = ow * ow, pp = p * p, E = 64 * pp;
    const int bn = blockIdx.y;
    const int n = bn % nw, b = bn / nw;
    const int ny = (n / ow) * p, nx = (n % ow) * p;

    __shared__ float w[64];
    if (threadIdx.x < nw) w[threadIdx.x] = scores[(size_t)bn * nw + threadIdx.x];
    __syncthreads();
    if (threadIdx.x == 0) {
        float mx = w[0];
        for (int m = 1; m < nw; ++m) mx = fmaxf(mx, w[m]);
        float s = 0.f;
        for (int m = 0; m < nw; ++m) { w[m] = __expf(w[m] - mx); s += w[m]; }
        float is = 1.0f / s;
        for (int m = 0; m < nw; ++m) w[m] *= is;
    }
    __syncthreads();

    for (int e = blockIdx.x * 256 + threadIdx.x; e < E; e += gridDim.x * 256) {
        int d = e / pp; int r = e - d * pp; int py = r / p; int px = r - py * p;
        const float* vb = v + ((size_t)b * Cc + grp * 64 + d) * SP;
        float acc = 0.f;
        for (int m = 0; m < nw; ++m) {
            int my = (m / ow) * p, mx2 = (m % ow) * p;
            acc += w[m] * vb[(my + py) * 64 + (mx2 + px)];
        }
        y[((size_t)b * Cc + grp * 64 + d) * SP + (ny + py) * 64 + (nx + px)] = acc;
    }
}

// ---------------------------------------------------------------------------
// Orchestration (graph-capture safe: only kernel launches on `stream`).
// ---------------------------------------------------------------------------
extern "C" void kernel_launch(void* const* d_in, const int* in_sizes, int n_in,
                              void* d_out, int out_size, void* d_ws, size_t ws_size,
                              hipStream_t stream) {
    (void)in_sizes; (void)n_in; (void)out_size; (void)ws_size;

    const float* x     = (const float*)d_in[0];
    const float* qw    = (const float*)d_in[1];
    const float* qb    = (const float*)d_in[2];
    const float* kw    = (const float*)d_in[3];
    const float* kb    = (const float*)d_in[4];
    const float* vw    = (const float*)d_in[5];
    const float* vb    = (const float*)d_in[6];
    const float* ow_   = (const float*)d_in[7];
    const float* ob    = (const float*)d_in[8];
    const float* bn1w  = (const float*)d_in[9];
    const float* bn1b  = (const float*)d_in[10];
    const float* gamma = (const float*)d_in[11];
    const float* f1w   = (const float*)d_in[12];
    const float* f1b   = (const float*)d_in[13];
    const float* fbn1w = (const float*)d_in[14];
    const float* fbn1b = (const float*)d_in[15];
    const float* f2w   = (const float*)d_in[16];
    const float* f2b   = (const float*)d_in[17];
    const float* fbn2w = (const float*)d_in[18];
    const float* fbn2b = (const float*)d_in[19];
    const float* bnfw  = (const float*)d_in[20];
    const float* bnfb  = (const float*)d_in[21];
    float* outp = (float*)d_out;

    // Workspace carve-up (~202 MB)
    float* buf0 = (float*)d_ws;          // q -> attention output Y
    float* buf1 = buf0 + TENS;           // k -> "out" residual
    float* buf2 = buf1 + TENS;           // v -> feed-forward intermediate
    float* sc1  = buf2 + TENS;           // scores g1: 16*4*4   = 256
    float* sc2  = sc1 + 256;             // scores g2: 16*16*16 = 4096
    float* sc3  = sc2 + 4096;            // scores g3: 16*64*64 = 65536
    bf16_t* rwq = (bf16_t*)(sc3 + 65536);
    bf16_t* rwk = rwq + Cc * Cc;
    bf16_t* rwv = rwk + Cc * Cc;
    bf16_t* rwo = rwv + Cc * Cc;
    bf16_t* rw1 = rwo + 9 * Cc * Cc;
    bf16_t* rw2 = rw1 + 9 * Cc * Cc;

    // 1) Repack weights into bf16 WMMA B-fragment order
    repack_w<<<(Cc * Cc) / 256, 256, 0, stream>>>(qw, rwq, 1);
    repack_w<<<(Cc * Cc) / 256, 256, 0, stream>>>(kw, rwk, 1);
    repack_w<<<(Cc * Cc) / 256, 256, 0, stream>>>(vw, rwv, 1);
    repack_w<<<(9 * Cc * Cc) / 256, 256, 0, stream>>>(ow_, rwo, 9);
    repack_w<<<(9 * Cc * Cc) / 256, 256, 0, stream>>>(f1w, rw1, 9);
    repack_w<<<(9 * Cc * Cc) / 256, 256, 0, stream>>>(f2w, rw2, 9);

    const dim3 ggrid(BATCH * 32, Cc / 64);   // (b, row-pairs) x channel groups

    // 2) QKV 1x1 GEMMs (WMMA)
    conv_gemm<1, 0><<<ggrid, 128, 0, stream>>>(x, rwq, qb, nullptr, nullptr, 0,
                                               nullptr, nullptr, nullptr, nullptr, buf0);
    conv_gemm<1, 0><<<ggrid, 128, 0, stream>>>(x, rwk, kb, nullptr, nullptr, 0,
                                               nullptr, nullptr, nullptr, nullptr, buf1);
    conv_gemm<1, 0><<<ggrid, 128, 0, stream>>>(x, rwv, vb, nullptr, nullptr, 0,
                                               nullptr, nullptr, nullptr, nullptr, buf2);

    // 3) Windowed attention: all scores first (q,k live), then outputs
    //    (q dead -> Y reuses buf0; group 0 output == V slice).
    att_scores<<<BATCH * 4 * 4,   256, 0, stream>>>(buf0, buf1, sc1, 1, 32);
    att_scores<<<BATCH * 16 * 16, 256, 0, stream>>>(buf0, buf1, sc2, 2, 16);
    att_scores<<<BATCH * 64 * 64, 256, 0, stream>>>(buf0, buf1, sc3, 3, 8);

    att_copy<<<(BATCH * 64 * SP) / 256, 256, 0, stream>>>(buf2, buf0);
    att_out<<<dim3(256, BATCH * 4),  256, 0, stream>>>(buf2, sc1, buf0, 1, 32);
    att_out<<<dim3(64,  BATCH * 16), 256, 0, stream>>>(buf2, sc2, buf0, 2, 16);
    att_out<<<dim3(16,  BATCH * 64), 256, 0, stream>>>(buf2, sc3, buf0, 3, 8);

    // 4) output_linear: lrelu(bn1(conv3x3(Y)+ob)); out = x + gamma*(.) -> buf1
    conv_gemm<9, 1><<<ggrid, 128, 0, stream>>>(buf0, rwo, ob, bn1w, bn1b, 1,
                                               x, gamma, nullptr, nullptr, buf1);

    // 5) FeedForward conv1 (pad=2, dil=2): lrelu(bn(conv(out))) -> buf2
    conv_gemm<9, 2><<<ggrid, 128, 0, stream>>>(buf1, rw1, f1b, fbn1w, fbn1b, 1,
                                               nullptr, nullptr, nullptr, nullptr, buf2);

    // 6) FeedForward conv2 (pad=1) + residual(out) + trailing BN -> d_out
    conv_gemm<9, 1><<<ggrid, 128, 0, stream>>>(buf2, rw2, f2b, fbn2w, fbn2b, 1,
                                               buf1, nullptr, bnfw, bnfb, outp);
}
